// GaussianImage_Cholesky_84456236908801
// MI455X (gfx1250) — matrix operand order repaired and checked by reading, not compiled, not run on hardware
//
#include <hip/hip_runtime.h>
#include <hip/hip_bf16.h>

typedef __attribute__((ext_vector_type(16))) _Float16 v16h;
typedef __attribute__((ext_vector_type(8)))  float    v8f;
typedef __attribute__((ext_vector_type(4)))  float    v4f;

#define NG     4096
#define H_IMG  512
#define W_IMG  512
#define HWPIX  (H_IMG * W_IMG)
#define MTILE  4            // 4 x 16-pixel strips per wave = 64 pixels/wave

#define ALPHA_MAXV 0.999f
#define ALPHA_MINV (1.0f / 255.0f)
#define LOG2E      1.442695040888963f

// One block = one image row (512 px). 8 waves/block, 64 px/wave.
// Inner loop: 32 Gaussians per step, alpha computed per-lane into f16
// A-fragments, accumulated via v_wmma_f32_16x16x32_f16 into a 16x16 f32 tile
// (16 pixels x 16 channels; ch0-2 rgb, ch3-5 geom, ch6 alpha-sum, ch7+ zero).
__global__ __launch_bounds__(256) void gsplat_wmma(
    const float* __restrict__ xyz,   // [N,2]
    const float* __restrict__ chol,  // [N,3]
    const float* __restrict__ feat,  // [N,3]
    const float* __restrict__ randc, // [N,3]
    float* __restrict__ out)         // [7*HW + N]
{
    // LDS staging: 64KB params4 + 16KB h2 + 64KB f16 color table = 144KB
    __shared__ v4f      s_p4[NG];        // (px, py, 0.5*c0, c1)
    __shared__ float    s_h2[NG];        // 0.5*c2
    __shared__ _Float16 s_col[8 * NG];   // channel-major: ch*NG + k; ch7 = 0

    const int tid = threadIdx.x;
    const int row = blockIdx.x;

    // ---- project Gaussians + build color table (cooperative, per block) ----
    for (int g = tid; g < NG; g += 256) {
        float mx = tanhf(xyz[2 * g + 0]);
        float my = tanhf(xyz[2 * g + 1]);
        float px = 0.5f * ((mx + 1.0f) * (float)W_IMG - 1.0f);
        float py = 0.5f * ((my + 1.0f) * (float)H_IMG - 1.0f);
        float l0 = chol[3 * g + 0] + 0.5f;
        float l1 = chol[3 * g + 1];
        float l2 = chol[3 * g + 2] + 0.5f;
        float s00 = l0 * l0;
        float s01 = l0 * l1;
        float s11 = l1 * l1 + l2 * l2;
        float inv = 1.0f / (s00 * s11 - s01 * s01);
        float c0 = s11 * inv;
        float c1 = -s01 * inv;
        float c2 = s00 * inv;
        s_p4[g] = (v4f){px, py, 0.5f * c0, c1};
        s_h2[g] = 0.5f * c2;
        s_col[0 * NG + g] = (_Float16)feat[3 * g + 0];
        s_col[1 * NG + g] = (_Float16)feat[3 * g + 1];
        s_col[2 * NG + g] = (_Float16)feat[3 * g + 2];
        s_col[3 * NG + g] = (_Float16)(0.5f * randc[3 * g + 0]);
        s_col[4 * NG + g] = (_Float16)(0.5f * randc[3 * g + 1]);
        s_col[5 * NG + g] = (_Float16)(0.5f * randc[3 * g + 2]);
        s_col[6 * NG + g] = (_Float16)1.0f;   // alpha-sum channel
        s_col[7 * NG + g] = (_Float16)0.0f;   // zero channel (read by lanes n>=8)
    }
    __syncthreads();

    const int lane   = tid & 31;
    const int wv     = tid >> 5;          // 0..7 -> 64-pixel segment
    const int n      = lane & 15;         // A row m / B,C column n
    const int koffA  = (lane >> 4) * 16;  // A-fragment K sub-offset (contiguous)
    const int koffB  = (lane >> 4) * 16;  // B-fragment K sub-offset (contiguous)
    const float pyf  = (float)row;

    // All lanes load a B fragment: lanes n>=8 read the zero channel (ch 7),
    // keeping the hot loop branch-free (EXEC stays all-ones throughout).
    const int cB = (n < 8) ? n : 7;
    const _Float16* __restrict__ colbase = &s_col[cB * NG + koffB];

    float pxf[MTILE];
#pragma unroll
    for (int s = 0; s < MTILE; ++s)
        pxf[s] = (float)(wv * 64 + s * 16 + n);

    v8f acc[MTILE];
#pragma unroll
    for (int s = 0; s < MTILE; ++s)
#pragma unroll
        for (int r = 0; r < 8; ++r)
            acc[s][r] = 0.0f;

    for (int kbase = 0; kbase < NG; kbase += 32) {
        // ---- B fragment: 32x16 f16, lane n = channel, contiguous K per half
        v16h bfrag = *(const v16h*)(colbase + kbase);   // 2x ds_load_b128

        // ---- A fragments: alpha(pixel m, gaussian k), contiguous K per half (same as B)
        v16h afrag[MTILE];
#pragma unroll
        for (int g = 0; g < 16; ++g) {
            int k = kbase + koffA + g;
            v4f   p  = s_p4[k];
            float h2 = s_h2[k];
            float dy   = pyf - p.y;
            float dyt  = h2 * dy * dy;
            float c1dy = p.w * dy;
#pragma unroll
            for (int s = 0; s < MTILE; ++s) {
                float dx    = pxf[s] - p.x;
                float sigma = fmaf(p.z * dx, dx, fmaf(c1dy, dx, dyt));
                float a     = __builtin_amdgcn_exp2f(-LOG2E * sigma);
                a = fminf(a, ALPHA_MAXV);
                a = (sigma < 0.0f || a < ALPHA_MINV) ? 0.0f : a;
                afrag[s][g] = (_Float16)a;
            }
        }

        // ---- 4 WMMAs share one B fragment (EXEC all-ones here)
#pragma unroll
        for (int s = 0; s < MTILE; ++s)
            acc[s] = __builtin_amdgcn_wmma_f32_16x16x32_f16(
                false, afrag[s], false, bfrag, (short)0, acc[s], false, false);
    }

    // ---- epilogue: C layout -> lane = channel n, VGPR r = pixel row m
    if (n < 7) {
        const int mhi = (lane >> 4) * 8;
#pragma unroll
        for (int s = 0; s < MTILE; ++s) {
#pragma unroll
            for (int r = 0; r < 8; ++r) {
                float v = acc[s][r];
                if (n < 6) v = fminf(fmaxf(v, 0.0f), 1.0f);  // clip rgb/geom
                int p = row * W_IMG + wv * 64 + s * 16 + r + mhi;
                out[n * HWPIX + p] = v;
            }
        }
    }
}

__global__ void fill_opac(float* __restrict__ out)
{
    int i = blockIdx.x * 256 + threadIdx.x;
    if (i < NG) out[7 * HWPIX + i] = 1.0f;
}

extern "C" void kernel_launch(void* const* d_in, const int* in_sizes, int n_in,
                              void* d_out, int out_size, void* d_ws, size_t ws_size,
                              hipStream_t stream)
{
    (void)in_sizes; (void)n_in; (void)d_ws; (void)ws_size; (void)out_size;
    const float* xyz   = (const float*)d_in[0];
    const float* chol  = (const float*)d_in[1];
    const float* feat  = (const float*)d_in[2];
    const float* randc = (const float*)d_in[3];
    float* out = (float*)d_out;

    gsplat_wmma<<<dim3(H_IMG), dim3(256), 0, stream>>>(xyz, chol, feat, randc, out);
    fill_opac<<<dim3((NG + 255) / 256), dim3(256), 0, stream>>>(out);
}
